// MS_Block_Conv_68607807586744
// MI455X (gfx1250) — matrix operand order repaired and checked by reading, not compiled, not run on hardware
//
#include <hip/hip_runtime.h>

// ---------------- problem constants ----------------
#define T_    4
#define B_    32
#define C_    512
#define N_    196              // H*W = 14*14
#define HID_  2048
#define M_    (T_ * B_ * N_)   // 25088 pixels
#define EPS_  1e-5f

typedef __attribute__((ext_vector_type(16))) __bf16 v16bf;
typedef __attribute__((ext_vector_type(8)))  float  v8f;

union FragBF { v16bf v; float4 f[2]; };

static __device__ __forceinline__ unsigned short f2bf(float f) {
  union { float f; unsigned u; } c; c.f = f;
  unsigned u = c.u;
  return (unsigned short)((u + 0x7FFFu + ((u >> 16) & 1u)) >> 16);
}

// ---------------- Tensor Data Mover support ----------------
#if defined(__has_builtin)
#if __has_builtin(__builtin_amdgcn_tensor_load_to_lds)
#define HAVE_TDM 1
#endif
#endif

#ifdef HAVE_TDM
typedef unsigned int u32x4 __attribute__((ext_vector_type(4)));
typedef int          i32x4 __attribute__((ext_vector_type(4)));
typedef int          i32x8 __attribute__((ext_vector_type(8)));

// Issue a TDM 2D tile load: tile_k (bf16 elems) x rows, row stride = strideK elems,
// packed contiguously into LDS at lds_off. D# layout per CDNA5 ISA.
static __device__ __forceinline__ void tdm_load_tile(
    unsigned lds_off, const unsigned short* g, int tile_k, int rows, int strideK)
{
  unsigned long long ga = (unsigned long long)(size_t)g;
  u32x4 g0;
  g0.x = 1u;                                                  // count=1, user mode
  g0.y = lds_off;                                             // lds_addr (bytes)
  g0.z = (unsigned)ga;                                        // global_addr[31:0]
  g0.w = ((unsigned)(ga >> 32) & 0x01FFFFFFu) | (2u << 30);   // addr[56:32] | type=2
  i32x8 g1;
  g1[0] = 1 << 16;          // workgroup_mask=0 | data_size=1 (2-byte)
  g1[1] = tile_k << 16;     // atomic_barrier_addr=0 | tensor_dim0[15:0]
  g1[2] = rows << 16;       // tensor_dim0[31:16]=0 | tensor_dim1[15:0]
  g1[3] = tile_k << 16;     // tensor_dim1[31:16]=0 | tile_dim0
  g1[4] = rows;             // tile_dim1 | tile_dim2=0
  g1[5] = strideK;          // tensor_dim0_stride[31:0]
  g1[6] = 0;                // stride hi | tensor_dim1_stride lo (unused, 2D)
  g1[7] = 0;
  i32x4 gz = {0, 0, 0, 0};
#if __clang_major__ >= 23
  i32x8 gz8 = {0, 0, 0, 0, 0, 0, 0, 0};
  __builtin_amdgcn_tensor_load_to_lds(g0, g1, gz, gz, gz8, 0);
#else
  __builtin_amdgcn_tensor_load_to_lds(g0, g1, gz, gz, 0);
#endif
}
#endif

// ---------------- bf16 WMMA GEMM: Y[M][Nout] = A[M][K] * W[Nout][K]^T ----------------
// 256 threads (8 waves). Block tile 128(M) x 128(N), K-step 64.
// Wave grid 4x2 -> wave tile 32x64 -> 2x4 v_wmma_f32_16x16x32_bf16 per K-substep.
// TDM double-buffers LDS: wave 0 issues descriptors, fences with s_wait_tensorcnt.
__global__ __launch_bounds__(256) void gemm_bf16_wmma(
    const unsigned short* __restrict__ A,   // [M][K] bf16 (spikes)
    const unsigned short* __restrict__ W,   // [Nout][K] bf16 (weights, native layout)
    float* __restrict__ Y,                  // [M][Nout] fp32
    int K, int Nout)
{
#ifdef HAVE_TDM
  __shared__ unsigned short As[2][128 * 64];
  __shared__ unsigned short Bs[2][128 * 64];
#else
  __shared__ unsigned short As1[128 * 64];
  __shared__ unsigned short Bs1[128 * 64];
#endif
  const int tid   = threadIdx.x;
  const int lane  = tid & 31;
  const int w     = tid >> 5;
  const int mBase = blockIdx.y * 128;
  const int nBase = blockIdx.x * 128;
  const int mW    = (w >> 1) * 32;
  const int nW    = (w & 1) * 64;
  const int hb    = (lane >> 4) * 8;    // A frag: K half-group base {0,8}
  const int kb    = (lane >> 4) * 8;    // B frag: K half-group base {0,8}

  v8f acc[2][4] = {};
  const int nk = K / 64;

#ifdef HAVE_TDM
  if (w == 0) {
    tdm_load_tile((unsigned)(size_t)&As[0][0], A + (size_t)mBase * K, 64, 128, K);
    tdm_load_tile((unsigned)(size_t)&Bs[0][0], W + (size_t)nBase * K, 64, 128, K);
  }
#endif

  for (int kt = 0; kt < nk; ++kt) {
#ifdef HAVE_TDM
    __syncthreads();   // all waves done reading the buffer TDM is about to overwrite
    if (w == 0) {
      if (kt + 1 < nk) {
        const int k0n = (kt + 1) * 64;
        tdm_load_tile((unsigned)(size_t)&As[(kt + 1) & 1][0],
                      A + (size_t)mBase * K + k0n, 64, 128, K);
        tdm_load_tile((unsigned)(size_t)&Bs[(kt + 1) & 1][0],
                      W + (size_t)nBase * K + k0n, 64, 128, K);
        __builtin_amdgcn_s_wait_tensorcnt(2);   // current buffer's 2 loads complete
      } else {
        __builtin_amdgcn_s_wait_tensorcnt(0);
      }
    }
    __syncthreads();   // publish current buffer to all waves
    const unsigned short* Ap = As[kt & 1];
    const unsigned short* Bp = Bs[kt & 1];
#else
    const int k0 = kt * 64;
    __syncthreads();
#pragma unroll
    for (int i = 0; i < 4; ++i) {
      int ch  = tid + 256 * i;          // 1024 chunks of 8 halves each, per matrix
      int row = ch >> 3;
      int ko  = (ch & 7) * 8;
      *(float4*)(&As1[row * 64 + ko]) =
          *(const float4*)(&A[(size_t)(mBase + row) * K + k0 + ko]);
      *(float4*)(&Bs1[row * 64 + ko]) =
          *(const float4*)(&W[(size_t)(nBase + row) * K + k0 + ko]);
    }
    __syncthreads();
    const unsigned short* Ap = As1;
    const unsigned short* Bp = Bs1;
#endif

#pragma unroll
    for (int ks = 0; ks < 2; ++ks) {
      FragBF a[2], b[4];
#pragma unroll
      for (int mt = 0; mt < 2; ++mt) {
        int r = mW + mt * 16 + (lane & 15);
        a[mt].f[0] = *(const float4*)(&Ap[r * 64 + ks * 32 + hb]);
        a[mt].f[1] = *(const float4*)(&Ap[r * 64 + ks * 32 + hb + 16]);
      }
#pragma unroll
      for (int nt = 0; nt < 4; ++nt) {
        int n = nW + nt * 16 + (lane & 15);
        b[nt].f[0] = *(const float4*)(&Bp[n * 64 + ks * 32 + kb]);
        b[nt].f[1] = *(const float4*)(&Bp[n * 64 + ks * 32 + kb + 16]);
      }
#pragma unroll
      for (int mt = 0; mt < 2; ++mt)
#pragma unroll
        for (int nt = 0; nt < 4; ++nt)
          acc[mt][nt] = __builtin_amdgcn_wmma_f32_16x16x32_bf16(
              false, a[mt].v, false, b[nt].v, (short)0, acc[mt][nt], false, false);
    }
  }

  // D layout: VGPR r holds M = r + (lane>=16 ? 8 : 0), N = lane&15
#pragma unroll
  for (int mt = 0; mt < 2; ++mt) {
    int rbase = mBase + mW + mt * 16 + ((lane >> 4) * 8);
#pragma unroll
    for (int nt = 0; nt < 4; ++nt) {
      int col = nBase + nW + nt * 16 + (lane & 15);
#pragma unroll
      for (int r = 0; r < 8; ++r)
        Y[(size_t)(rbase + r) * Nout + col] = acc[mt][nt][r];
    }
  }
}

// ---------------- weight fp32 -> bf16 pack ----------------
__global__ void cvt_bf16(const float* __restrict__ w, unsigned short* __restrict__ o, int n) {
  int i = blockIdx.x * blockDim.x + threadIdx.x;
  if (i < n) o[i] = f2bf(w[i]);
}

// ---------------- LayerNorm over C, channel-major input [T,B,C,N] -> pixel-major [M][C] ----------------
__global__ __launch_bounds__(256) void ln_chmajor(
    const float* __restrict__ x, const float* __restrict__ g,
    const float* __restrict__ bb, float* __restrict__ out)
{
  __shared__ float sS[256], sQ[256];
  int p = blockIdx.x;
  int tb = p / N_, n = p % N_;
  const float* base = x + (size_t)tb * C_ * N_ + n;
  float s = 0.f, s2 = 0.f;
  for (int c = threadIdx.x; c < C_; c += 256) {
    float v = base[(size_t)c * N_];
    s += v; s2 += v * v;
  }
  sS[threadIdx.x] = s; sQ[threadIdx.x] = s2;
  __syncthreads();
  for (int off = 128; off > 0; off >>= 1) {
    if (threadIdx.x < off) { sS[threadIdx.x] += sS[threadIdx.x + off]; sQ[threadIdx.x] += sQ[threadIdx.x + off]; }
    __syncthreads();
  }
  float mu  = sS[0] / (float)C_;
  float var = sQ[0] / (float)C_ - mu * mu;
  float inv = rsqrtf(var + EPS_);
  for (int c = threadIdx.x; c < C_; c += 256) {
    float v = base[(size_t)c * N_];
    out[(size_t)p * C_ + c] = (v - mu) * inv * g[c] + bb[c];
  }
}

// ---------------- LayerNorm over C, pixel-major input/output [M][C] ----------------
__global__ __launch_bounds__(256) void ln_pixmajor(
    const float* __restrict__ x, const float* __restrict__ g,
    const float* __restrict__ bb, float* __restrict__ out)
{
  __shared__ float sS[256], sQ[256];
  int p = blockIdx.x;
  const float* base = x + (size_t)p * C_;
  float s = 0.f, s2 = 0.f;
  for (int c = threadIdx.x; c < C_; c += 256) {
    float v = base[c];
    s += v; s2 += v * v;
  }
  sS[threadIdx.x] = s; sQ[threadIdx.x] = s2;
  __syncthreads();
  for (int off = 128; off > 0; off >>= 1) {
    if (threadIdx.x < off) { sS[threadIdx.x] += sS[threadIdx.x + off]; sQ[threadIdx.x] += sQ[threadIdx.x + off]; }
    __syncthreads();
  }
  float mu  = sS[0] / (float)C_;
  float var = sQ[0] / (float)C_ - mu * mu;
  float inv = rsqrtf(var + EPS_);
  for (int c = threadIdx.x; c < C_; c += 256) {
    float v = base[c];
    out[(size_t)p * C_ + c] = (v - mu) * inv * g[c] + bb[c];
  }
}

// ---------------- LIF over T (fp32 in, bf16 spikes out), v_th = 1 ----------------
__global__ void lif_t(const float* __restrict__ xin, unsigned short* __restrict__ spk) {
  int i = blockIdx.x * blockDim.x + threadIdx.x;           // over B*N*C
  if (i >= B_ * N_ * C_) return;
  int c = i % C_; int rest = i / C_; int n = rest % N_; int b = rest / N_;
  float v = 0.f;
#pragma unroll
  for (int t = 0; t < T_; ++t) {
    size_t p = ((size_t)(t * B_ + b) * N_ + n) * C_ + c;
    v = 0.5f * (v + xin[p]);
    bool sp = v >= 1.0f;
    spk[p] = sp ? (unsigned short)0x3F80 : (unsigned short)0;
    if (sp) v = 0.f;
  }
}

// ---------------- BN affine + LIF over T (GEMM out -> spikes), v_th = 1 ----------------
__global__ void bn_lif(const float* __restrict__ y, const float* __restrict__ scale,
                       const float* __restrict__ shift, unsigned short* __restrict__ spk, int Cw) {
  long long total = (long long)B_ * N_ * Cw;
  long long i = (long long)blockIdx.x * blockDim.x + threadIdx.x;
  if (i >= total) return;
  int c = (int)(i % Cw); long long rest = i / Cw;
  int n = (int)(rest % N_); int b = (int)(rest / N_);
  float sc = scale[c], sh = shift[c];
  float v = 0.f;
#pragma unroll
  for (int t = 0; t < T_; ++t) {
    size_t p = ((size_t)(t * B_ + b) * N_ + n) * Cw + c;
    float xv = y[p] * sc + sh;
    v = 0.5f * (v + xv);
    bool sp = v >= 1.0f;
    spk[p] = sp ? (unsigned short)0x3F80 : (unsigned short)0;
    if (sp) v = 0.f;
  }
}

// ---------------- BN statistics ----------------
__global__ void zero_f(float* a, int n) {
  int i = blockIdx.x * blockDim.x + threadIdx.x;
  if (i < n) a[i] = 0.f;
}

__global__ void stats_partial(const float* __restrict__ y, float* __restrict__ sum,
                              float* __restrict__ sq, int Nout) {
  int c = blockIdx.x * blockDim.x + threadIdx.x;   // gridDim.x*256 == Nout
  float s = 0.f, s2 = 0.f;
  for (int r = blockIdx.y; r < M_; r += gridDim.y) {
    float v = y[(size_t)r * Nout + c];
    s += v; s2 += v * v;
  }
  atomicAdd(&sum[c], s);
  atomicAdd(&sq[c], s2);
}

__global__ void bn_finalize(const float* __restrict__ sum, const float* __restrict__ sq,
                            const float* __restrict__ g, const float* __restrict__ bb,
                            float* __restrict__ scale, float* __restrict__ shift, int Cw) {
  int c = blockIdx.x * blockDim.x + threadIdx.x;
  if (c >= Cw) return;
  float mean = sum[c] / (float)M_;
  float var  = sq[c] / (float)M_ - mean * mean;
  float sc   = g[c] * rsqrtf(var + EPS_);
  scale[c] = sc;
  shift[c] = bb[c] - mean * sc;
}

// ---------------- kv = lif(sum_n k*v, v_th=0.5) per (t,b,c) ----------------
__global__ void kv_reduce_lif(const unsigned short* __restrict__ ks,
                              const unsigned short* __restrict__ vs,
                              unsigned short* __restrict__ kvspk) {
  int i = blockIdx.x * blockDim.x + threadIdx.x;   // B*C
  if (i >= B_ * C_) return;
  int c = i % C_, b = i / C_;
  float v = 0.f;
  for (int t = 0; t < T_; ++t) {
    float s = 0.f;
    size_t base = (size_t)(t * B_ + b) * N_ * C_ + c;
    for (int n = 0; n < N_; ++n)
      s += (ks[base + (size_t)n * C_] && vs[base + (size_t)n * C_]) ? 1.f : 0.f;
    v = 0.5f * (v + s);
    bool sp = v >= 0.5f;
    kvspk[(size_t)(t * B_ + b) * C_ + c] = sp ? (unsigned short)1 : (unsigned short)0;
    if (sp) v = 0.f;
  }
}

// ---------------- xo = q * kv (binary product, bf16 spikes) ----------------
__global__ void xo_mul(const unsigned short* __restrict__ qs,
                       const unsigned short* __restrict__ kvspk,
                       unsigned short* __restrict__ s) {
  int i = blockIdx.x * blockDim.x + threadIdx.x;   // M*C
  if (i >= M_ * C_) return;
  int c = i % C_; int p = i / C_; int tb = p / N_;
  s[i] = (qs[i] && kvspk[(size_t)tb * C_ + c]) ? (unsigned short)0x3F80 : (unsigned short)0;
}

// ---------------- BN affine + residual add in place: a = y*sc+sh + a ----------------
__global__ void bn_add(const float* __restrict__ y, const float* __restrict__ scale,
                       const float* __restrict__ shift, float* __restrict__ a) {
  int i = blockIdx.x * blockDim.x + threadIdx.x;
  if (i >= M_ * C_) return;
  int c = i % C_;
  a[i] = y[i] * scale[c] + shift[c] + a[i];
}

// ---------------- final: out[T,B,C,N] = BN(yo) + xm  (pixel-major -> channel-major) ----------------
__global__ void bn_add_out(const float* __restrict__ yo, const float* __restrict__ scale,
                           const float* __restrict__ shift, const float* __restrict__ xm,
                           float* __restrict__ out) {
  int i = blockIdx.x * blockDim.x + threadIdx.x;   // channel-major flat index
  if (i >= M_ * C_) return;
  int n = i % N_; int c = (i / N_) % C_; int tb = i / (N_ * C_);
  size_t pm = (size_t)(tb * N_ + n) * C_ + c;
  out[i] = yo[pm] * scale[c] + shift[c] + xm[pm];
}

// =====================================================================
extern "C" void kernel_launch(void* const* d_in, const int* in_sizes, int n_in,
                              void* d_out, int out_size, void* d_ws, size_t ws_size,
                              hipStream_t stream) {
  const float* x     = (const float*)d_in[0];
  const float* ln1_g = (const float*)d_in[1];
  const float* ln1_b = (const float*)d_in[2];
  const float* ln2_g = (const float*)d_in[3];
  const float* ln2_b = (const float*)d_in[4];
  const float* wq    = (const float*)d_in[5];
  const float* bnq_g = (const float*)d_in[6];
  const float* bnq_b = (const float*)d_in[7];
  const float* wk    = (const float*)d_in[8];
  const float* bnk_g = (const float*)d_in[9];
  const float* bnk_b = (const float*)d_in[10];
  const float* wv    = (const float*)d_in[11];
  const float* bnv_g = (const float*)d_in[12];
  const float* bnv_b = (const float*)d_in[13];
  const float* wp    = (const float*)d_in[14];
  // d_in[15] = bp  : absorbed by BN mean subtraction (no-op)
  const float* bnp_g = (const float*)d_in[16];
  const float* bnp_b = (const float*)d_in[17];
  const float* w1    = (const float*)d_in[18];
  // d_in[19] = b1  : absorbed by BN
  const float* bn1_g = (const float*)d_in[20];
  const float* bn1_b = (const float*)d_in[21];
  const float* w2    = (const float*)d_in[22];
  // d_in[23] = b2  : absorbed by BN
  const float* bn2_g = (const float*)d_in[24];
  const float* bn2_b = (const float*)d_in[25];

  // ---- carve workspace ----
  char* wsp = (char*)d_ws;
  auto carve = [&](size_t bytes) -> char* {
    char* r = wsp;
    wsp += (bytes + 255) & ~(size_t)255;
    return r;
  };
  float*          A    = (float*)carve((size_t)M_ * C_ * 4);      // xa / identity1 / x2 / yo
  float*          Bb   = (float*)carve((size_t)M_ * C_ * 4);      // gemm out / xm
  unsigned short* S    = (unsigned short*)carve((size_t)M_ * C_ * 2);   // spikes (reused)
  unsigned short* Qs   = (unsigned short*)carve((size_t)M_ * C_ * 2);
  unsigned short* Ks   = (unsigned short*)carve((size_t)M_ * C_ * 2);
  unsigned short* Vs   = (unsigned short*)carve((size_t)M_ * C_ * 2);
  float*          Hf   = (float*)carve((size_t)M_ * HID_ * 4);    // MLP hidden gemm out
  unsigned short* HS   = (unsigned short*)carve((size_t)M_ * HID_ * 2);
  unsigned short* Wqb  = (unsigned short*)carve((size_t)C_ * C_ * 2);
  unsigned short* Wkb  = (unsigned short*)carve((size_t)C_ * C_ * 2);
  unsigned short* Wvb  = (unsigned short*)carve((size_t)C_ * C_ * 2);
  unsigned short* Wpb  = (unsigned short*)carve((size_t)C_ * C_ * 2);
  unsigned short* W1b  = (unsigned short*)carve((size_t)HID_ * C_ * 2);
  unsigned short* W2b  = (unsigned short*)carve((size_t)C_ * HID_ * 2);
  unsigned short* KV   = (unsigned short*)carve((size_t)T_ * B_ * C_ * 2);
  float*          stat = (float*)carve((size_t)2 * HID_ * 4);     // [sum | sumsq]
  float*          scl  = (float*)carve((size_t)HID_ * 4);
  float*          shf  = (float*)carve((size_t)HID_ * 4);
  (void)ws_size; (void)in_sizes; (void)n_in; (void)out_size;

  const int TPB  = 256;
  const int gMC  = (M_ * C_ + TPB - 1) / TPB;          // 50176
  const int gBNC = (B_ * N_ * C_ + TPB - 1) / TPB;     // 12544
  const int gBNH = (B_ * N_ * HID_ + TPB - 1) / TPB;   // 50176

  // 0. pack weights fp32 -> bf16 (native [Co][Ci] layout)
  cvt_bf16<<<(C_ * C_ + TPB - 1) / TPB, TPB, 0, stream>>>(wq, Wqb, C_ * C_);
  cvt_bf16<<<(C_ * C_ + TPB - 1) / TPB, TPB, 0, stream>>>(wk, Wkb, C_ * C_);
  cvt_bf16<<<(C_ * C_ + TPB - 1) / TPB, TPB, 0, stream>>>(wv, Wvb, C_ * C_);
  cvt_bf16<<<(C_ * C_ + TPB - 1) / TPB, TPB, 0, stream>>>(wp, Wpb, C_ * C_);
  cvt_bf16<<<(HID_ * C_ + TPB - 1) / TPB, TPB, 0, stream>>>(w1, W1b, HID_ * C_);
  cvt_bf16<<<(C_ * HID_ + TPB - 1) / TPB, TPB, 0, stream>>>(w2, W2b, C_ * HID_);

  // 1. LN1 (channel-major in -> pixel-major xa); keep as identity1 in A
  ln_chmajor<<<M_, TPB, 0, stream>>>(x, ln1_g, ln1_b, A);
  // 2. LIF over T -> spikes S
  lif_t<<<gBNC, TPB, 0, stream>>>(A, S);

  // 3. q/k/v branches: GEMM -> BN stats -> BN+LIF spikes
  const unsigned short* Wb3[3] = {Wqb, Wkb, Wvb};
  const float* g3[3] = {bnq_g, bnk_g, bnv_g};
  const float* b3[3] = {bnq_b, bnk_b, bnv_b};
  unsigned short* spk3[3] = {Qs, Ks, Vs};
  for (int br = 0; br < 3; ++br) {
    zero_f<<<(2 * HID_ + TPB - 1) / TPB, TPB, 0, stream>>>(stat, 2 * HID_);
    gemm_bf16_wmma<<<dim3(C_ / 128, M_ / 128), TPB, 0, stream>>>(S, Wb3[br], Bb, C_, C_);
    stats_partial<<<dim3(C_ / TPB, 64), TPB, 0, stream>>>(Bb, stat, stat + HID_, C_);
    bn_finalize<<<(C_ + TPB - 1) / TPB, TPB, 0, stream>>>(stat, stat + HID_, g3[br], b3[br], scl, shf, C_);
    bn_lif<<<gBNC, TPB, 0, stream>>>(Bb, scl, shf, spk3[br], C_);
  }

  // 4. kv = lif(sum_n k*v, 0.5); 5. xo = q*kv -> S (binary)
  kv_reduce_lif<<<(B_ * C_ + TPB - 1) / TPB, TPB, 0, stream>>>(Ks, Vs, KV);
  xo_mul<<<gMC, TPB, 0, stream>>>(Qs, KV, S);

  // 6. projection GEMM + BN + residual (into A, in place)
  zero_f<<<(2 * HID_ + TPB - 1) / TPB, TPB, 0, stream>>>(stat, 2 * HID_);
  gemm_bf16_wmma<<<dim3(C_ / 128, M_ / 128), TPB, 0, stream>>>(S, Wpb, Bb, C_, C_);
  stats_partial<<<dim3(C_ / TPB, 64), TPB, 0, stream>>>(Bb, stat, stat + HID_, C_);
  bn_finalize<<<(C_ + TPB - 1) / TPB, TPB, 0, stream>>>(stat, stat + HID_, bnp_g, bnp_b, scl, shf, C_);
  bn_add<<<gMC, TPB, 0, stream>>>(Bb, scl, shf, A);      // A = x2

  // 7. LN2 (pixel-major): A -> Bb (xm = identity2); 8. LIF -> S
  ln_pixmajor<<<M_, TPB, 0, stream>>>(A, ln2_g, ln2_b, Bb);
  lif_t<<<gBNC, TPB, 0, stream>>>(Bb, S);

  // 9. MLP up: C -> HID
  zero_f<<<(2 * HID_ + TPB - 1) / TPB, TPB, 0, stream>>>(stat, 2 * HID_);
  gemm_bf16_wmma<<<dim3(HID_ / 128, M_ / 128), TPB, 0, stream>>>(S, W1b, Hf, C_, HID_);
  stats_partial<<<dim3(HID_ / TPB, 64), TPB, 0, stream>>>(Hf, stat, stat + HID_, HID_);
  bn_finalize<<<(HID_ + TPB - 1) / TPB, TPB, 0, stream>>>(stat, stat + HID_, bn1_g, bn1_b, scl, shf, HID_);
  bn_lif<<<gBNH, TPB, 0, stream>>>(Hf, scl, shf, HS, HID_);

  // 10. MLP down: HID -> C, BN, + identity2, transpose to [T,B,C,H,W]
  zero_f<<<(2 * HID_ + TPB - 1) / TPB, TPB, 0, stream>>>(stat, 2 * HID_);
  gemm_bf16_wmma<<<dim3(C_ / 128, M_ / 128), TPB, 0, stream>>>(HS, W2b, A, HID_, C_);
  stats_partial<<<dim3(C_ / TPB, 64), TPB, 0, stream>>>(A, stat, stat + HID_, C_);
  bn_finalize<<<(C_ + TPB - 1) / TPB, TPB, 0, stream>>>(stat, stat + HID_, bn2_g, bn2_b, scl, shf, C_);
  bn_add_out<<<gMC, TPB, 0, stream>>>(A, scl, shf, Bb, (float*)d_out);
}
